// GlobalAttention_5257039970385
// MI455X (gfx1250) — matrix operand order repaired and checked by reading, not compiled, not run on hardware
//
#include <hip/hip_runtime.h>
#include <cstdint>
#include <cstddef>

#define D_DIM 1024
#define S_DIM 2048
#define B_DIM 64

typedef __attribute__((ext_vector_type(16))) __bf16 v16bf;
typedef __attribute__((ext_vector_type(8)))  float  v8f;
typedef __attribute__((ext_vector_type(4)))  unsigned int u32x4;
typedef __attribute__((ext_vector_type(2)))  unsigned int u32x2;
typedef __attribute__((ext_vector_type(4)))  float  f32x4;

__device__ __forceinline__ unsigned short f2bf(float x) {
    union { float f; unsigned u; } v; v.f = x;
    unsigned r = v.u + 0x7FFFu + ((v.u >> 16) & 1u);   // round-to-nearest-even
    return (unsigned short)(r >> 16);
}

__device__ __forceinline__ void wait_async0() {
#if __has_builtin(__builtin_amdgcn_s_wait_asynccnt)
    __builtin_amdgcn_s_wait_asynccnt(0);
#else
    asm volatile("s_wait_asynccnt 0" ::: "memory");
#endif
}

// ---------------------------------------------------------------- K0: W_ctx -> bf16
__global__ void k_prep_wctx(const float* __restrict__ W, unsigned short* __restrict__ Wb) {
    int i = blockIdx.x * blockDim.x + threadIdx.x;      // i in [0, D*D/4)
    f32x4 w = ((const f32x4*)W)[i];
    u32x2 p;
    p.x = (unsigned)f2bf(w.x) | ((unsigned)f2bf(w.y) << 16);
    p.y = (unsigned)f2bf(w.z) | ((unsigned)f2bf(w.w) << 16);
    ((u32x2*)Wb)[i] = p;
}

// ---------------------------------------------------------------- K1: target = input @ W_in.T
__global__ void k_target(const float* __restrict__ in, const float* __restrict__ Win,
                         float* __restrict__ tgt) {
    int e = blockIdx.x * 256 + threadIdx.x;
    int b = blockIdx.y;
    const float* ir = in + (size_t)b * D_DIM;
    const float* wr = Win + (size_t)e * D_DIM;
    float s = 0.f;
    for (int k = 0; k < D_DIM; k += 4) {
        f32x4 iv = *(const f32x4*)(ir + k);
        f32x4 wv = *(const f32x4*)(wr + k);
        s += iv.x * wv.x + iv.y * wv.y + iv.z * wv.z + iv.w * wv.w;
    }
    tgt[(size_t)b * D_DIM + e] = s;
}

// ---------------------------------------------------------------- K2 helpers
// Async-copy one 32-wide K chunk of bf16 W_ctx (1024 rows x 32 halfs) into LDS,
// pitch 40 halfs. 512 threads x 8 transfers x 16 B, tracked by ASYNCcnt.
__device__ __forceinline__ void stage_b_async(unsigned long long wbBase, int kc,
                                              unsigned short* bbuf, int t) {
#pragma unroll
    for (int i = 0; i < 8; ++i) {
        int linear = i * 512 + t;
        int row = linear >> 2, seg = linear & 3;
        unsigned ldsAddr = (unsigned)(size_t)&bbuf[row * 40 + seg * 8];
        unsigned gOff    = (unsigned)((row * D_DIM + kc + seg * 8) * 2);
        asm volatile("global_load_async_to_lds_b128 %0, %1, %2"
                     :: "v"(ldsAddr), "v"(gOff), "s"(wbBase)
                     : "memory");
    }
}

// Stage the 64x32 f32 context tile as bf16 into LDS (needs VALU conversion).
__device__ __forceinline__ void stage_a(const float* __restrict__ ctx, size_t ctxRow,
                                        int kc, unsigned short* abuf, int rowA, int segA) {
    f32x4 v = *(const f32x4*)(ctx + ctxRow + kc + segA * 4);
    u32x2 p;
    p.x = (unsigned)f2bf(v.x) | ((unsigned)f2bf(v.y) << 16);
    p.y = (unsigned)f2bf(v.z) | ((unsigned)f2bf(v.w) << 16);
    *(u32x2*)&abuf[rowA * 40 + segA * 4] = p;
}

// ---------------------------------------------------------------- K2: logits via WMMA bf16
// grid (S/64, B), 512 threads = 16 waves = 4 m-tiles x 4 n-groups.
// Double-buffered LDS; next chunk's B slab arrives via async global->LDS DMA
// while current chunk's 16 WMMAs run.
__launch_bounds__(512, 1)
__global__ void k_logits(const float* __restrict__ ctx,
                         const unsigned short* __restrict__ Wb,
                         const float* __restrict__ tgt,
                         const float* __restrict__ Wone,
                         const float* __restrict__ bone,
                         float* __restrict__ logits) {
    __shared__ __align__(16) unsigned short ldsA[2][64 * 40];     // 2 x  5,120 B
    __shared__ __align__(16) unsigned short ldsB[2][1024 * 40];   // 2 x 81,920 B
    __shared__ float ldsRed[64 * 4];

    const int t      = threadIdx.x;
    const int b      = blockIdx.y;
    const int sBlock = blockIdx.x * 64;
    const int wave   = t >> 5;
    const int lane   = t & 31;
    const int half   = lane >> 4;
    const int l16    = lane & 15;
    const int mtile  = wave & 3;    // which 16-row slab of s
    const int ngrp   = wave >> 2;   // which 256-col slab of d

    const unsigned long long wbBase = (unsigned long long)(size_t)Wb;

    v8f c[16];
#pragma unroll
    for (int s = 0; s < 16; ++s) c[s] = (v8f){0.f,0.f,0.f,0.f,0.f,0.f,0.f,0.f};

    const int rowA = t >> 3, segA = t & 7;
    const size_t ctxRow = ((size_t)b * S_DIM + sBlock + rowA) * D_DIM;

    // Prologue: stage chunk 0 into buffer 0
    stage_b_async(wbBase, 0, ldsB[0], t);
    stage_a(ctx, ctxRow, 0, ldsA[0], rowA, segA);
    wait_async0();
    __syncthreads();

    for (int kc = 0; kc < D_DIM; kc += 32) {
        const int cur = (kc >> 5) & 1;
        // Prefetch next chunk into the other buffer (overlaps with WMMAs below)
        if (kc + 32 < D_DIM) {
            stage_b_async(wbBase, kc + 32, ldsB[cur ^ 1], t);
            stage_a(ctx, ctxRow, kc + 32, ldsA[cur ^ 1], rowA, segA);
        }

        // A fragment (16x32 bf16): lane row = l16, K 0-15 / 16-31 split on lane>>4
        const unsigned short* pA = &ldsA[cur][(mtile * 16 + l16) * 40 + half * 16];
        union { u32x4 q[2]; v16bf v; } ua;
        ua.q[0] = *(const u32x4*)pA;
        ua.q[1] = *(const u32x4*)(pA + 8);

#pragma unroll
        for (int s = 0; s < 16; ++s) {
            // B fragment (32x16 bf16): lane col = l16, K 0-15 / 16-31 split on lane>>4
            const unsigned short* pB = &ldsB[cur][(ngrp * 256 + s * 16 + l16) * 40 + half * 16];
            union { u32x4 q[2]; v16bf v; } ub;
            ub.q[0] = *(const u32x4*)pB;
            ub.q[1] = *(const u32x4*)(pB + 8);
            c[s] = __builtin_amdgcn_wmma_f32_16x16x32_bf16(
                       false, ua.v, false, ub.v, (short)0, c[s], false, false);
        }

        wait_async0();       // next-chunk DMA complete (no-op on last iteration)
        __syncthreads();     // everyone done reading cur + writing nxt
    }

    // Epilogue: logit partial = sum_n Wone[n] * tanh(C[m,n] + target[b,n])
    float acc[8];
#pragma unroll
    for (int r = 0; r < 8; ++r) acc[r] = 0.f;
#pragma unroll
    for (int s = 0; s < 16; ++s) {
        int n = ngrp * 256 + s * 16 + l16;
        float tg = tgt[(size_t)b * D_DIM + n];
        float wo = Wone[n];
#pragma unroll
        for (int r = 0; r < 8; ++r) {
            float v = wo * tanhf(c[s][r] + tg);
            v += __shfl_xor(v, 1, 16);
            v += __shfl_xor(v, 2, 16);
            v += __shfl_xor(v, 4, 16);
            v += __shfl_xor(v, 8, 16);
            acc[r] += v;                 // C layout: VGPR r -> row r + 8*half
        }
    }
    if (l16 == 0) {
#pragma unroll
        for (int r = 0; r < 8; ++r)
            ldsRed[(mtile * 16 + half * 8 + r) * 4 + ngrp] = acc[r];
    }
    __syncthreads();
    if (t < 64) {
        float s = bone[0] + ldsRed[t*4+0] + ldsRed[t*4+1] + ldsRed[t*4+2] + ldsRed[t*4+3];
        logits[(size_t)b * S_DIM + sBlock + t] = s;
    }
}

// ---------------------------------------------------------------- K3: softmax over S (+mask)
__global__ void k_softmax(const float* __restrict__ logits,
                          const unsigned char* __restrict__ mask,
                          float* __restrict__ attn) {
    __shared__ float red[256];
    int b = blockIdx.x, t = threadIdx.x;
    float l[8]; float mx = -INFINITY;
#pragma unroll
    for (int i = 0; i < 8; ++i) {
        int s = t + i * 256;
        float v = logits[(size_t)b * S_DIM + s];
        if (mask[(size_t)b * S_DIM + s]) v = -INFINITY;
        l[i] = v; mx = fmaxf(mx, v);
    }
    red[t] = mx; __syncthreads();
    for (int o = 128; o > 0; o >>= 1) { if (t < o) red[t] = fmaxf(red[t], red[t+o]); __syncthreads(); }
    mx = red[0]; __syncthreads();
    float sum = 0.f;
#pragma unroll
    for (int i = 0; i < 8; ++i) {
        l[i] = (l[i] == -INFINITY) ? 0.f : __expf(l[i] - mx);
        sum += l[i];
    }
    red[t] = sum; __syncthreads();
    for (int o = 128; o > 0; o >>= 1) { if (t < o) red[t] += red[t+o]; __syncthreads(); }
    float inv = 1.f / red[0];
#pragma unroll
    for (int i = 0; i < 8; ++i)
        attn[(size_t)b * S_DIM + t + i * 256] = l[i] * inv;
}

// ---------------------------------------------------------------- K4: weighted_ctx = attn . context
__global__ void k_wctx(const float* __restrict__ ctx, const float* __restrict__ attn,
                       float* __restrict__ wc) {
    __shared__ float a[S_DIM];
    int b = blockIdx.y;
    int d = blockIdx.x * 256 + threadIdx.x;
    for (int i = threadIdx.x; i < S_DIM; i += 256) a[i] = attn[(size_t)b * S_DIM + i];
    __syncthreads();
    float acc = 0.f;
    const float* cp = ctx + (size_t)b * S_DIM * D_DIM + d;
    for (int s = 0; s < S_DIM; ++s) acc += a[s] * cp[(size_t)s * D_DIM];
    wc[(size_t)b * D_DIM + d] = acc;
}

// ---------------------------------------------------------------- K5: gate + gated vector
__global__ void k_gate(const float* __restrict__ wc, const float* __restrict__ in,
                       const float* __restrict__ Wcg, const float* __restrict__ bcg,
                       float* __restrict__ gated) {
    int b = blockIdx.y;
    int w = threadIdx.x >> 5, lane = threadIdx.x & 31;
    int d = blockIdx.x * 8 + w;
    const float* row = Wcg + (size_t)d * (2 * D_DIM);
    const float* wcb = wc + (size_t)b * D_DIM;
    const float* inb = in + (size_t)b * D_DIM;
    float s = 0.f;
    for (int k = lane; k < D_DIM; k += 32) s += row[k] * wcb[k];
    for (int k = lane; k < D_DIM; k += 32) s += row[D_DIM + k] * inb[k];
    for (int m = 16; m; m >>= 1) s += __shfl_xor(s, m, 32);
    if (lane == 0) {
        float g = 1.f / (1.f + __expf(-(s + bcg[d])));
        gated[(size_t)b * 2 * D_DIM + d]         = wcb[d] * g;
        gated[(size_t)b * 2 * D_DIM + D_DIM + d] = inb[d] * (1.f - g);
    }
}

// ---------------------------------------------------------------- K6: out = tanh(gated @ W_out.T)
__global__ void k_out(const float* __restrict__ gated, const float* __restrict__ Wout,
                      float* __restrict__ out) {
    int b = blockIdx.y;
    int w = threadIdx.x >> 5, lane = threadIdx.x & 31;
    int e = blockIdx.x * 8 + w;
    const float* row = Wout + (size_t)e * (2 * D_DIM);
    const float* g   = gated + (size_t)b * (2 * D_DIM);
    float s = 0.f;
    for (int k = lane; k < 2 * D_DIM; k += 32) s += row[k] * g[k];
    for (int m = 16; m; m >>= 1) s += __shfl_xor(s, m, 32);
    if (lane == 0) out[(size_t)b * D_DIM + e] = tanhf(s);
}

// ----------------------------------------------------------------
extern "C" void kernel_launch(void* const* d_in, const int* in_sizes, int n_in,
                              void* d_out, int out_size, void* d_ws, size_t ws_size,
                              hipStream_t stream) {
    const float*         input = (const float*)d_in[0];
    const float*         ctx   = (const float*)d_in[1];
    const unsigned char* mask  = (const unsigned char*)d_in[2];
    const float*         Win   = (const float*)d_in[3];
    const float*         Wctx  = (const float*)d_in[4];
    const float*         Wone  = (const float*)d_in[5];
    const float*         bone  = (const float*)d_in[6];
    const float*         Wcg   = (const float*)d_in[7];
    const float*         bcg   = (const float*)d_in[8];
    const float*         Wout  = (const float*)d_in[9];

    float* out  = (float*)d_out;                         // [B, D]
    float* attn = (float*)d_out + (size_t)B_DIM * D_DIM; // [B, S]

    char* ws = (char*)d_ws;
    unsigned short* Wb  = (unsigned short*)(ws);                         // 2 MB bf16 W_ctx
    float* tgt    = (float*)(ws + 2*1024*1024);                          // B*D
    float* logits = (float*)(ws + 2*1024*1024 + 256*1024);               // B*S
    float* wc     = (float*)(ws + 2*1024*1024 + 256*1024 + 512*1024);    // B*D
    float* gated  = (float*)(ws + 2*1024*1024 + 256*1024 + 512*1024 + 256*1024); // B*2D

    k_prep_wctx<<<dim3((D_DIM * D_DIM / 4) / 256), 256, 0, stream>>>(Wctx, Wb);
    k_target  <<<dim3(D_DIM / 256, B_DIM), 256, 0, stream>>>(input, Win, tgt);
    k_logits  <<<dim3(S_DIM / 64, B_DIM), 512, 0, stream>>>(ctx, Wb, tgt, Wone, bone, logits);
    k_softmax <<<dim3(B_DIM), 256, 0, stream>>>(logits, mask, attn);
    k_wctx    <<<dim3(D_DIM / 256, B_DIM), 256, 0, stream>>>(ctx, attn, wc);
    k_gate    <<<dim3(D_DIM / 8, B_DIM), 256, 0, stream>>>(wc, input, Wcg, bcg, gated);
    k_out     <<<dim3(D_DIM / 8, B_DIM), 256, 0, stream>>>(gated, Wout, out);
}
